// MultiHeadDenseAttention_13280038879454
// MI455X (gfx1250) — compile-verified, matrix operand order checked by reading, not run on hardware
//
#include <hip/hip_runtime.h>
#include <math.h>

// ---------------- CDNA5 WMMA types ----------------
typedef __attribute__((ext_vector_type(16))) __bf16 v16bf;
typedef __attribute__((ext_vector_type(8)))  float  v8f;
typedef unsigned short u16;
typedef unsigned int   u32;

struct alignas(16) U128 { u32 x, y, z, w; };
union FragU { v16bf v; U128 u[2]; };

// Problem constants
#define BATCH    2
#define NHEAD    16
#define SEQ      2048
#define HDIM     64
#define FEAT     1024
#define MROWS    (BATCH * SEQ)   // 4096

__device__ __forceinline__ u16 f32_to_bf16(float f) {
  u32 u = __float_as_uint(f);
  u32 r = (u + 0x7FFFu + ((u >> 16) & 1u)) >> 16;   // round-to-nearest-even
  return (u16)r;
}

__device__ __forceinline__ v8f wmma_bf16(v16bf a, v16bf b, v8f c) {
  // v_wmma_f32_16x16x32_bf16: D = A(16x32) * B(32x16) + C(16x16 f32)
  return __builtin_amdgcn_wmma_f32_16x16x32_bf16(
      /*neg_a=*/false, a, /*neg_b=*/false, b,
      /*c_mod=*/(short)0, c, /*reuse_a=*/false, /*reuse_b=*/false);
}

// Load a 16x32 bf16 fragment from a row-major [16 x >=k0+32] tile.
// Per-lane layout (ISA 7.12.2, 16-bit A 16x32): lane L (L<16): row M=L,
// K = lk..lk+7 (VGPR0-3) and lk+16..lk+23 (VGPR4-7), lk = (L>>4)*8.
// Same pattern serves B when the B operand is stored as Bt row-major [N x K].
__device__ __forceinline__ v16bf load_frag(const u16* tile, int ld, int k0, int lane) {
  const int lr = lane & 15;
  const int lk = (lane >> 4) << 3;
  const u16* p = tile + (size_t)lr * ld + (size_t)(k0 + lk);
  FragU f;
  f.u[0] = *(const U128*)p;          // K = k0+lk   .. +7
  f.u[1] = *(const U128*)(p + 16);   // K = k0+lk+16.. +23
  return f.v;
}

// ---------------- fp32 -> bf16 convert ----------------
__global__ void __launch_bounds__(256) cvt_bf16_kernel(
    const float* __restrict__ in, u16* __restrict__ out, int n) {
  int i = blockIdx.x * 256 + threadIdx.x;
  if (i < n) out[i] = f32_to_bf16(in[i]);
}

// ---------------- generic GEMM: C[16x64 tile] = A @ Bt^T ----------------
// A: [M x K] bf16 row-major, Bt: [N x K] bf16 row-major.
// EPI==0: store fp32 row-major to Cf[M x N]
// EPI==1: store bf16 per-head transposed to CvT[b][h][d][m]  (value path)
template <int EPI>
__global__ void __launch_bounds__(128) gemm16x64_bf16(
    const u16* __restrict__ A, const u16* __restrict__ Bt,
    float* __restrict__ Cf, u16* __restrict__ CvT,
    int M, int N, int K) {
  const int lane = threadIdx.x & 31;
  const int wid  = threadIdx.x >> 5;
  const int tile = blockIdx.x * 4 + wid;
  const int tilesN = N >> 6;
  const int tm = tile / tilesN;
  const int tn = tile - tm * tilesN;
  const int M0 = tm << 4, N0 = tn << 6;
  const int lr = lane & 15, lk = (lane >> 4) << 3;

  v8f acc[4] = {};
  const u16* Abase = A + (size_t)M0 * K;
  for (int k0 = 0; k0 < K; k0 += 32) {
    __builtin_prefetch(Abase + (size_t)lr * K + k0 + 128, 0, 1);  // global_prefetch_b8
    v16bf a = load_frag(Abase, K, k0, lane);
#pragma unroll
    for (int nt = 0; nt < 4; ++nt) {
      v16bf bfrag = load_frag(Bt + (size_t)(N0 + nt * 16) * K, K, k0, lane);
      acc[nt] = wmma_bf16(a, bfrag, acc[nt]);
    }
  }
#pragma unroll
  for (int nt = 0; nt < 4; ++nt) {
#pragma unroll
    for (int r = 0; r < 8; ++r) {
      const int grow = M0 + r + lk;          // D frag: M = r + 8*(lane>=16)
      const int gcol = N0 + nt * 16 + lr;    //         N = lane&15
      if (EPI == 0) {
        Cf[(size_t)grow * N + gcol] = acc[nt][r];
      } else {
        const int bb = grow >> 11, m = grow & (SEQ - 1);
        const int hh = gcol >> 6,  dd = gcol & (HDIM - 1);
        CvT[((size_t)((bb << 4) + hh) * HDIM + dd) * SEQ + m] = f32_to_bf16(acc[nt][r]);
      }
    }
  }
}

// ---------------- fused per-head MLP-attention (flash style) ----------------
// Per wave: one 16-query tile of one (b,h).
//   hid  = relu(x_h[16x64] @ W1^T + b1)           (4x2 WMMA, LDS re-layout)
//   loop 32 keys: S = hid @ W2chunk^T + b2        (4 WMMA)
//                 online softmax (shfl_xor within 16-lane halves)
//                 P -> LDS -> A-frag; O += P @ Vt (4 WMMA)
__global__ void __launch_bounds__(128) attn_fused_kernel(
    const u16* __restrict__ xb, const u16* __restrict__ W1b,
    const float* __restrict__ b1, const u16* __restrict__ W2b,
    const float* __restrict__ b2, const u16* __restrict__ vTb,
    u16* __restrict__ outb) {
  __shared__ __align__(16) u16 hid_s[4][16 * HDIM];  // 2KB / wave
  __shared__ __align__(16) u16 p_s[4][16 * 32];      // 1KB / wave

  const int lane = threadIdx.x & 31;
  const int wid  = threadIdx.x >> 5;
  const int tile = blockIdx.x * 4 + wid;
  const int tile_n = tile & 127;       // 128 query tiles per head
  const int bh = tile >> 7;
  const int h = bh & (NHEAD - 1);
  const int b = bh >> 4;
  const int row0 = tile_n << 4;
  const int lr = lane & 15;
  const int lk = (lane >> 4) << 3;

  // ---- hid = relu(x_h @ W1^T + b1) ----
  const u16* xbase = xb + ((size_t)(b * SEQ + row0)) * FEAT + h * HDIM;
  v16bf ax0 = load_frag(xbase, FEAT, 0, lane);
  v16bf ax1 = load_frag(xbase, FEAT, 32, lane);
#pragma unroll
  for (int nt = 0; nt < 4; ++nt) {
    v16bf bw0 = load_frag(W1b + (size_t)(nt * 16) * HDIM, HDIM, 0, lane);
    v16bf bw1 = load_frag(W1b + (size_t)(nt * 16) * HDIM, HDIM, 32, lane);
    v8f c = {};
    c = wmma_bf16(ax0, bw0, c);
    c = wmma_bf16(ax1, bw1, c);
    const float bias = b1[nt * 16 + lr];
#pragma unroll
    for (int r = 0; r < 8; ++r) {
      float v = c[r] + bias;
      v = v > 0.f ? v : 0.f;                               // relu
      hid_s[wid][(r + lk) * HDIM + nt * 16 + lr] = f32_to_bf16(v);
    }
  }
  asm volatile("s_wait_dscnt 0" ::: "memory");             // D-layout -> A-layout via LDS
  v16bf ah0 = load_frag(&hid_s[wid][0], HDIM, 0, lane);
  v16bf ah1 = load_frag(&hid_s[wid][0], HDIM, 32, lane);

  v8f vacc[4] = {};
  float rmax[8], rsum[8];
#pragma unroll
  for (int r = 0; r < 8; ++r) { rmax[r] = -1e30f; rsum[r] = 0.f; }

  const u16* vbase = vTb + ((size_t)((b * NHEAD + h) * HDIM)) * SEQ;

  for (int m0 = 0; m0 < SEQ; m0 += 32) {
    // S tiles (two 16x16), K-dim = head dim 64
    v8f s0, s1;
    {
      const u16* wb = W2b + (size_t)m0 * HDIM;
      v8f c = {};
      c = wmma_bf16(ah0, load_frag(wb, HDIM, 0, lane), c);
      c = wmma_bf16(ah1, load_frag(wb, HDIM, 32, lane), c);
      s0 = c + b2[m0 + lr];
    }
    {
      const u16* wb = W2b + (size_t)(m0 + 16) * HDIM;
      v8f c = {};
      c = wmma_bf16(ah0, load_frag(wb, HDIM, 0, lane), c);
      c = wmma_bf16(ah1, load_frag(wb, HDIM, 32, lane), c);
      s1 = c + b2[m0 + 16 + lr];
    }
    // online softmax: rows live in fixed vgpr index within each 16-lane half
#pragma unroll
    for (int r = 0; r < 8; ++r) {
      float v = fmaxf(s0[r], s1[r]);
      v = fmaxf(v, __shfl_xor(v, 1));
      v = fmaxf(v, __shfl_xor(v, 2));
      v = fmaxf(v, __shfl_xor(v, 4));
      v = fmaxf(v, __shfl_xor(v, 8));
      const float nm = fmaxf(rmax[r], v);
      const float sc = __expf(rmax[r] - nm);
      rmax[r] = nm;
      rsum[r] *= sc;
      vacc[0][r] *= sc; vacc[1][r] *= sc; vacc[2][r] *= sc; vacc[3][r] *= sc;
      const float p0 = __expf(s0[r] - nm);
      const float p1 = __expf(s1[r] - nm);
      rsum[r] += p0 + p1;                     // per-lane partial; reduce at end
      p_s[wid][(r + lk) * 32 + lr]      = f32_to_bf16(p0);
      p_s[wid][(r + lk) * 32 + 16 + lr] = f32_to_bf16(p1);
    }
    asm volatile("s_wait_dscnt 0" ::: "memory");
    v16bf ap = load_frag(&p_s[wid][0], 32, 0, lane);       // P as 16x32 A-frag
#pragma unroll
    for (int nt = 0; nt < 4; ++nt) {                       // O += P @ VtChunk^T
      v16bf bv = load_frag(vbase + (size_t)(nt * 16) * SEQ, SEQ, m0, lane);
      vacc[nt] = wmma_bf16(ap, bv, vacc[nt]);
    }
  }

  // epilogue: finish row sums, normalize, store merged-head bf16
#pragma unroll
  for (int r = 0; r < 8; ++r) {
    float v = rsum[r];
    v += __shfl_xor(v, 1);
    v += __shfl_xor(v, 2);
    v += __shfl_xor(v, 4);
    v += __shfl_xor(v, 8);
    const float inv = 1.f / v;
#pragma unroll
    for (int nt = 0; nt < 4; ++nt) {
      const int grow = b * SEQ + row0 + r + lk;
      const int gcol = h * HDIM + nt * 16 + lr;
      outb[(size_t)grow * FEAT + gcol] = f32_to_bf16(vacc[nt][r] * inv);
    }
  }
}

// ---------------- workspace layout (bytes) ----------------
static constexpr size_t OFF_XB   = 0;                       // x bf16   [4096x1024]  8 MB
static constexpr size_t OFF_WVB  = OFF_XB   + (size_t)MROWS * FEAT * 2;
static constexpr size_t OFF_WOB  = OFF_WVB  + (size_t)FEAT * FEAT * 2;
static constexpr size_t OFF_W1B  = OFF_WOB  + (size_t)FEAT * FEAT * 2;
static constexpr size_t OFF_W2B  = OFF_W1B  + (size_t)HDIM * HDIM * 2;
static constexpr size_t OFF_VT   = OFF_W2B  + (size_t)SEQ * HDIM * 2;   // Vt bf16 [b][h][d][m] 8 MB
static constexpr size_t OFF_ATTN = OFF_VT   + (size_t)MROWS * FEAT * 2; // attn out bf16 8 MB

extern "C" void kernel_launch(void* const* d_in, const int* in_sizes, int n_in,
                              void* d_out, int out_size, void* d_ws, size_t ws_size,
                              hipStream_t stream) {
  const float* x  = (const float*)d_in[0];
  const float* W1 = (const float*)d_in[1];
  const float* b1 = (const float*)d_in[2];
  const float* W2 = (const float*)d_in[3];
  const float* b2 = (const float*)d_in[4];
  const float* Wv = (const float*)d_in[5];
  const float* Wo = (const float*)d_in[6];
  float* out = (float*)d_out;

  char* ws = (char*)d_ws;
  u16* xb    = (u16*)(ws + OFF_XB);
  u16* Wvb   = (u16*)(ws + OFF_WVB);
  u16* Wob   = (u16*)(ws + OFF_WOB);
  u16* W1b   = (u16*)(ws + OFF_W1B);
  u16* W2b   = (u16*)(ws + OFF_W2B);
  u16* vT    = (u16*)(ws + OFF_VT);
  u16* attnb = (u16*)(ws + OFF_ATTN);

  const int n_x  = MROWS * FEAT;      // 4194304
  const int n_w  = FEAT * FEAT;       // 1048576
  const int n_w1 = HDIM * HDIM;       // 4096
  const int n_w2 = SEQ * HDIM;        // 131072
  cvt_bf16_kernel<<<(n_x  + 255) / 256, 256, 0, stream>>>(x,  xb,  n_x);
  cvt_bf16_kernel<<<(n_w  + 255) / 256, 256, 0, stream>>>(Wv, Wvb, n_w);
  cvt_bf16_kernel<<<(n_w  + 255) / 256, 256, 0, stream>>>(Wo, Wob, n_w);
  cvt_bf16_kernel<<<(n_w1 + 255) / 256, 256, 0, stream>>>(W1, W1b, n_w1);
  cvt_bf16_kernel<<<(n_w2 + 255) / 256, 256, 0, stream>>>(W2, W2b, n_w2);

  // value = x @ Wv^T, stored transposed per head as Vt (bf16)
  const int gemm_blocks = (MROWS / 16) * (FEAT / 64) / 4;   // 1024
  gemm16x64_bf16<1><<<gemm_blocks, 128, 0, stream>>>(xb, Wvb, nullptr, vT,
                                                     MROWS, FEAT, FEAT);

  // fused hid/logits/softmax/PV : one 16-query tile per wave
  const int attn_blocks = (BATCH * NHEAD * (SEQ / 16)) / 4; // 1024
  attn_fused_kernel<<<attn_blocks, 128, 0, stream>>>(xb, W1b, b1, W2b, b2, vT, attnb);

  // out = attn_merged @ Wo^T (fp32 out)
  gemm16x64_bf16<0><<<gemm_blocks, 128, 0, stream>>>(attnb, Wob, out, nullptr,
                                                     MROWS, FEAT, FEAT);
}
